// Model_39676907884687
// MI455X (gfx1250) — matrix operand order repaired and checked by reading, not compiled, not run on hardware
//
#include <hip/hip_runtime.h>

// ---------- types ----------
typedef __attribute__((ext_vector_type(16))) __bf16 bf16x16;
typedef __attribute__((ext_vector_type(8)))  __bf16 bf16x8;
typedef __attribute__((ext_vector_type(8)))  float  v8f;

// ---------- f32 -> bf16 (round to nearest even) ----------
__device__ __forceinline__ __bf16 f2bf(float f) {
  unsigned u = __builtin_bit_cast(unsigned, f);
  unsigned r = u + 0x7FFFu + ((u >> 16) & 1u);
  unsigned short h = (unsigned short)(r >> 16);
  return __builtin_bit_cast(__bf16, h);
}

// ---------- WMMA fragment loads (wave32, 16x16x32 bf16) ----------
// A (16 rows x 32 K), row-major source with leading dim `ld` (elements).
// Lane L: row = L&15; lanes 0-15 hold K {0..7,16..23}, lanes 16-31 hold K {8..15,24..31}.
__device__ __forceinline__ bf16x16 load_frag_a(const __bf16* __restrict__ base,
                                               long long ld, int lane) {
  int row = lane & 15, hi = lane >> 4;
  const __bf16* p = base + (long long)row * ld + hi * 8;
  union { bf16x16 v; bf16x8 h[2]; } u;
  u.h[0] = *(const bf16x8*)(p);
  u.h[1] = *(const bf16x8*)(p + 16);
  return u.v;
}

// B (32 K x 16 cols); source holds B^T rows (i.e. column n of B = row n of source).
// Lane L: col = L&15; lanes 0-15 hold K {0..7,16..23}, lanes 16-31 hold K {8..15,24..31}.
__device__ __forceinline__ bf16x16 load_frag_b(const __bf16* __restrict__ base,
                                               long long ld, int lane) {
  int col = lane & 15, hi = lane >> 4;
  const __bf16* p = base + (long long)col * ld + hi * 8;
  union { bf16x16 v; bf16x8 h[2]; } u;
  u.h[0] = *(const bf16x8*)(p);
  u.h[1] = *(const bf16x8*)(p + 16);
  return u.v;
}

// ---------- one-wave 64x64 bf16 GEMM core: C += A(64xK) * B(Kx64) ----------
// A: row-major, lda elements. Bt: rows are columns of B (row-major, ldb elements).
__device__ __forceinline__ void wave_gemm_64x64(const __bf16* __restrict__ A, long long lda,
                                                const __bf16* __restrict__ Bt, long long ldb,
                                                int K, v8f acc[4][4], int lane) {
#pragma unroll 1
  for (int k0 = 0; k0 < K; k0 += 32) {
    bf16x16 af[4], bfr[4];
#pragma unroll
    for (int i = 0; i < 4; ++i)
      af[i] = load_frag_a(A + (long long)(16 * i) * lda + k0, lda, lane);
#pragma unroll
    for (int j = 0; j < 4; ++j)
      bfr[j] = load_frag_b(Bt + (long long)(16 * j) * ldb + k0, ldb, lane);
#pragma unroll
    for (int i = 0; i < 4; ++i)
#pragma unroll
      for (int j = 0; j < 4; ++j)
        acc[i][j] = __builtin_amdgcn_wmma_f32_16x16x32_bf16(
            false, af[i], false, bfr[j], (short)0, acc[i][j], false, false);
  }
}

// ---------- elementwise f32 -> bf16 conversion ----------
__global__ __launch_bounds__(256) void cvt_f32_bf16(const float* __restrict__ src,
                                                    __bf16* __restrict__ dst, long long n) {
  long long i = (long long)blockIdx.x * blockDim.x + threadIdx.x;
  long long stride = (long long)gridDim.x * blockDim.x;
  for (; i < n; i += stride) dst[i] = f2bf(src[i]);
}

// ---------- fused QKV projection: qkv = x @ W^T + b ----------
// M = B*S = 16384, N = 3072, K = 1024. One wave per 64x64 tile.
// Writes q,k row-major bf16 [B*S,1024]; v transposed bf16 [B][1024][4096].
__global__ __launch_bounds__(32) void qkv_gemm_kernel(const __bf16* __restrict__ xbf,
                                                      const __bf16* __restrict__ wbf,
                                                      const float* __restrict__ bias,
                                                      __bf16* __restrict__ qbf,
                                                      __bf16* __restrict__ kbf,
                                                      __bf16* __restrict__ vT) {
  int lane = threadIdx.x;
  int n0 = blockIdx.x * 64;
  int m0 = blockIdx.y * 64;
  v8f acc[4][4] = {};
  wave_gemm_64x64(xbf + (long long)m0 * 1024, 1024,
                  wbf + (long long)n0 * 1024, 1024, 1024, acc, lane);
  int col = lane & 15, hi = lane >> 4;
#pragma unroll
  for (int i = 0; i < 4; ++i)
#pragma unroll
    for (int j = 0; j < 4; ++j) {
      int n = n0 + 16 * j + col;
      float bv = bias[n];
#pragma unroll
      for (int r = 0; r < 8; ++r) {
        int m = m0 + 16 * i + r + 8 * hi;
        __bf16 o = f2bf(acc[i][j][r] + bv);
        if (n < 1024) {
          qbf[(long long)m * 1024 + n] = o;
        } else if (n < 2048) {
          kbf[(long long)m * 1024 + (n - 1024)] = o;
        } else {
          int b = m >> 12, s = m & 4095;
          vT[((long long)b * 1024 + (n - 2048)) * 4096 + s] = o;
        }
      }
    }
}

// ---------- scores = scale * q @ k^T (per batch), f32 out [4096,4096] ----------
__global__ __launch_bounds__(32) void scores_kernel(const __bf16* __restrict__ q,
                                                    const __bf16* __restrict__ k,
                                                    float* __restrict__ Sc, float scale) {
  int lane = threadIdx.x;
  int n0 = blockIdx.x * 64;
  int m0 = blockIdx.y * 64;
  v8f acc[4][4] = {};
  wave_gemm_64x64(q + (long long)m0 * 1024, 1024,
                  k + (long long)n0 * 1024, 1024, 1024, acc, lane);
  int col = lane & 15, hi = lane >> 4;
#pragma unroll
  for (int i = 0; i < 4; ++i)
#pragma unroll
    for (int j = 0; j < 4; ++j)
#pragma unroll
      for (int r = 0; r < 8; ++r) {
        int m = m0 + 16 * i + r + 8 * hi;
        int n = n0 + 16 * j + col;
        Sc[(long long)m * 4096 + n] = acc[i][j][r] * scale;
      }
}

// ---------- row softmax: f32 scores [4096,4096] -> bf16 probs ----------
__global__ __launch_bounds__(256) void softmax_kernel(const float* __restrict__ Sc,
                                                      __bf16* __restrict__ P) {
  __shared__ float red[256];
  int row = blockIdx.x;
  int tid = threadIdx.x;
  const float* s = Sc + (long long)row * 4096;
  float vals[16];
  float m = -3.4e38f;
#pragma unroll
  for (int i = 0; i < 16; ++i) {
    vals[i] = s[tid + i * 256];
    m = fmaxf(m, vals[i]);
  }
  red[tid] = m;
  __syncthreads();
  for (int w = 128; w > 0; w >>= 1) {
    if (tid < w) red[tid] = fmaxf(red[tid], red[tid + w]);
    __syncthreads();
  }
  m = red[0];
  __syncthreads();
  float sum = 0.f;
#pragma unroll
  for (int i = 0; i < 16; ++i) {
    vals[i] = __expf(vals[i] - m);
    sum += vals[i];
  }
  red[tid] = sum;
  __syncthreads();
  for (int w = 128; w > 0; w >>= 1) {
    if (tid < w) red[tid] += red[tid + w];
    __syncthreads();
  }
  float inv = 1.0f / red[0];
  __bf16* p = P + (long long)row * 4096;
#pragma unroll
  for (int i = 0; i < 16; ++i) p[tid + i * 256] = f2bf(vals[i] * inv);
}

// ---------- out = P @ V (per batch): A = P bf16 [4096,4096], B cols = vT rows ----------
__global__ __launch_bounds__(32) void pv_kernel(const __bf16* __restrict__ P,
                                                const __bf16* __restrict__ vT,
                                                float* __restrict__ out) {
  int lane = threadIdx.x;
  int n0 = blockIdx.x * 64;  // d  (0..1023)
  int m0 = blockIdx.y * 64;  // query (0..4095)
  v8f acc[4][4] = {};
  wave_gemm_64x64(P + (long long)m0 * 4096, 4096,
                  vT + (long long)n0 * 4096, 4096, 4096, acc, lane);
  int col = lane & 15, hi = lane >> 4;
#pragma unroll
  for (int i = 0; i < 4; ++i)
#pragma unroll
    for (int j = 0; j < 4; ++j)
#pragma unroll
      for (int r = 0; r < 8; ++r) {
        int m = m0 + 16 * i + r + 8 * hi;
        int n = n0 + 16 * j + col;
        out[(long long)m * 1024 + n] = acc[i][j][r];
      }
}

extern "C" void kernel_launch(void* const* d_in, const int* in_sizes, int n_in,
                              void* d_out, int out_size, void* d_ws, size_t ws_size,
                              hipStream_t stream) {
  (void)in_sizes; (void)n_in; (void)out_size; (void)ws_size;
  const float* x    = (const float*)d_in[0];  // [4,4096,1024]
  const float* W    = (const float*)d_in[1];  // [3072,1024]
  const float* bias = (const float*)d_in[2];  // [3072]
  float* out        = (float*)d_out;          // [4,4096,1024]

  const long long S = 4096, D = 1024;
  const long long xN = 4 * S * D;      // 16,777,216
  const long long wN = 3 * D * D;      // 3,145,728

  char* ws = (char*)d_ws;
  size_t off = 0;
  auto take = [&](size_t bytes) -> char* {
    char* p = ws + off;
    off += (bytes + 255) & ~(size_t)255;
    return p;
  };
  __bf16* xbf = (__bf16*)take(xN * 2);      // 33.5 MB
  __bf16* wbf = (__bf16*)take(wN * 2);      //  6.3 MB
  __bf16* qbf = (__bf16*)take(xN * 2);      // 33.5 MB
  __bf16* kbf = (__bf16*)take(xN * 2);      // 33.5 MB
  __bf16* vT  = (__bf16*)take(xN * 2);      // 33.5 MB (transposed per batch)
  float*  Sb  = (float*)take(S * S * 4);    // 67.1 MB (reused per batch)
  __bf16* Pb  = (__bf16*)take(S * S * 2);   // 33.5 MB (reused per batch)

  // 1) fp32 -> bf16 staging
  cvt_f32_bf16<<<2048, 256, 0, stream>>>(x, xbf, xN);
  cvt_f32_bf16<<<1024, 256, 0, stream>>>(W, wbf, wN);

  // 2) fused QKV projection (M=16384, N=3072, K=1024)
  qkv_gemm_kernel<<<dim3(48, 256), 32, 0, stream>>>(xbf, wbf, bias, qbf, kbf, vT);

  // 3) attention, batch-looped (score buffer reused; stream order gives deps)
  const float scale = 0.03125f;  // 1024^-0.5
  for (int b = 0; b < 4; ++b) {
    const __bf16* qb = qbf + (long long)b * S * D;
    const __bf16* kb = kbf + (long long)b * S * D;
    const __bf16* vb = vT  + (long long)b * D * S;
    float* ob = out + (long long)b * S * D;
    scores_kernel<<<dim3(64, 64), 32, 0, stream>>>(qb, kb, Sb, scale);
    softmax_kernel<<<dim3((int)S), 256, 0, stream>>>(Sb, Pb);
    pv_kernel<<<dim3(16, 64), 32, 0, stream>>>(Pb, vb, ob);
  }
}
